// SwinTransformerBlock_63969242907185
// MI455X (gfx1250) — matrix operand order repaired and checked by reading, not compiled, not run on hardware
//
#include <hip/hip_runtime.h>
#include <math.h>

// ---------------------------------------------------------------------------
// Types matching gfx1250 WMMA builtin signatures
// ---------------------------------------------------------------------------
typedef __attribute__((ext_vector_type(16))) __bf16 v16bf;
typedef __attribute__((ext_vector_type(8)))  __bf16 v8bf;
typedef __attribute__((ext_vector_type(8)))  float  v8f;

#define SHUF16 0,1,2,3,4,5,6,7,8,9,10,11,12,13,14,15

// A-matrix fragment (16x32 bf16, row-major source, leading dim ld).
// ISA layout: lanes0-15 row m=lane, elems0-7 = K k..k+7, elems8-15 = K k+16..k+23
//             lanes16-31 row m=lane-16, K shifted by +8 / +24.
__device__ __forceinline__ v16bf load_a_frag(const __bf16* base, int ld, int k, int lane) {
  const __bf16* p = base + (size_t)(lane & 15) * ld + k + ((lane >> 4) << 3);
  v8bf lo = *(const v8bf*)p;
  v8bf hi = *(const v8bf*)(p + 16);
  return __builtin_shufflevector(lo, hi, SHUF16);
}

// B-matrix fragment (32x16 bf16) where B[k][n] = W[n][k], W row-major (N,K):
// lane n<16 -> column n, elems0-7 = K k..k+7, elems8-15 = K k+16..k+23,
// lane n>=16 -> column n-16, K shifted by +8 / +24.
__device__ __forceinline__ v16bf load_b_frag(const __bf16* base, int ld, int k, int lane) {
  const __bf16* p = base + (size_t)(lane & 15) * ld + k + ((lane >> 4) << 3);
  v8bf lo = *(const v8bf*)p;
  v8bf hi = *(const v8bf*)(p + 16);
  return __builtin_shufflevector(lo, hi, SHUF16);
}

__device__ __forceinline__ v8f wmma_bf16(v16bf a, v16bf b, v8f c) {
  return __builtin_amdgcn_wmma_f32_16x16x32_bf16(false, a, false, b, (short)0, c, false, false);
}

// gfx1250 async global->LDS copies (ASYNCcnt) when the toolchain exposes them.
// Builtin signature (from hipcc diagnostic): (v4i addrspace(1)*, v4i addrspace(3)*,
// imm int offset, imm int cpol) with GCC-style vector pointee.
#if __has_builtin(__builtin_amdgcn_global_load_async_to_lds_b128) && \
    __has_builtin(__builtin_amdgcn_s_wait_asynccnt)
#define USE_ASYNC_LDS 1
typedef int v4i_g __attribute__((vector_size(16)));
typedef __attribute__((address_space(1))) v4i_g GV4;
typedef __attribute__((address_space(3))) v4i_g LV4;
__device__ __forceinline__ void async_cp16(const __bf16* g, __bf16* l) {
  __builtin_amdgcn_global_load_async_to_lds_b128((GV4*)g, (LV4*)l, 0, 0);
}
#else
#define USE_ASYNC_LDS 0
#endif

// ---------------------------------------------------------------------------
// Static problem configuration
// ---------------------------------------------------------------------------
// B=4, H=W=128, C=768, NH=12, HD=64, window 8x8 (NT=64), shift 4
#define TOK_TOTAL 65536
#define CCH       768
#define C3        2304
#define CHID      3072
#define LDS_LD    40          // padded leading dim (80B = 20 banks; conflict-free)

// ---------------------------------------------------------------------------
// Epilogue functors
// ---------------------------------------------------------------------------
struct EpiBias {               // out[m*ldo+n] = bf16(v + bias[n])
  const float* bias; __bf16* out; int ldo;
  __device__ void operator()(int m, int n, float v) const {
    out[(size_t)m * ldo + n] = (__bf16)(v + bias[n]);
  }
};

struct EpiGelu {               // exact-gelu epilogue for fc1
  const float* bias; __bf16* out; int ldo;
  __device__ void operator()(int m, int n, float v) const {
    float xv = v + bias[n];
    float g  = 0.5f * xv * (1.0f + erff(xv * 0.7071067811865476f));
    out[(size_t)m * ldo + n] = (__bf16)g;
  }
};

struct EpiProj {               // proj: window-reverse + unshift scatter + residual
  const float* bias; const float* shortcut; float* out;
  __device__ void operator()(int m, int n, float v) const {
    int wdx = m >> 6, t = m & 63;
    int bi = wdx >> 8, wl = wdx & 255;
    int hs = (wl >> 4) * 8 + (t >> 3);
    int ws = (wl & 15) * 8 + (t & 7);
    int h  = (hs + 4) & 127, wp = (ws + 4) & 127;
    size_t d = ((size_t)bi * 16384 + (size_t)h * 128 + wp) * CCH + n;
    out[d] = shortcut[d] + v + bias[n];
  }
};

struct EpiAdd {                // fc2: accumulate into f32 residual stream
  const float* bias; float* out;
  __device__ void operator()(int m, int n, float v) const {
    out[(size_t)m * CCH + n] += v + bias[n];
  }
};

// ---------------------------------------------------------------------------
// bf16 GEMM: D[m][n] = sum_k A[m][k] * W[n][k]  (+ epilogue)
// Block tile 128(M) x 128(N), K-step 32, double-buffered LDS staging.
// 8 waves arranged 4(M) x 2(N); each wave computes 32x64 via eight
// 16x16x32 WMMAs per K-step (2 A frags x 4 B frags from LDS).
// Staging: 4 x B128 per thread per K-step (async global->LDS on gfx1250
// toolchains that expose the builtin, else global->VGPR->ds_store with the
// stores deferred past the WMMAs so the loads overlap compute).
// ---------------------------------------------------------------------------
template <class Epi>
__global__ __launch_bounds__(256)
void gemm_bf16(const __bf16* __restrict__ A, const __bf16* __restrict__ W,
               int K, int lda, int ldw, Epi epi) {
  __shared__ __bf16 As[2][128 * LDS_LD];
  __shared__ __bf16 Bs[2][128 * LDS_LD];

  const int tid  = threadIdx.x;
  const int lane = tid & 31;
  const int wave = tid >> 5;
  const int wm = wave & 3, wn = wave >> 2;
  const int m0 = blockIdx.x * 128;
  const int n0 = blockIdx.y * 128;
  const __bf16* Ag = A + (size_t)m0 * lda;
  const __bf16* Wg = W + (size_t)n0 * ldw;

  // staging chunks: thread covers rows r0 and r0+64, 8-elem segment s0
  const int r0 = tid >> 2;
  const int s0 = (tid & 3) * 8;
  const int lo = r0 * LDS_LD + s0;          // LDS element offsets
  const int hi = (r0 + 64) * LDS_LD + s0;

#if USE_ASYNC_LDS
  auto stage = [&](int kk, int buf) {
    async_cp16(Ag + (size_t)r0 * lda + kk + s0,        &As[buf][lo]);
    async_cp16(Ag + (size_t)(r0 + 64) * lda + kk + s0, &As[buf][hi]);
    async_cp16(Wg + (size_t)r0 * ldw + kk + s0,        &Bs[buf][lo]);
    async_cp16(Wg + (size_t)(r0 + 64) * ldw + kk + s0, &Bs[buf][hi]);
  };
#endif

  v8f acc0[4] = {}, acc1[4] = {};

  // ---- preload first K tile
#if USE_ASYNC_LDS
  stage(0, 0);
  __builtin_amdgcn_s_wait_asynccnt(0);
#else
  {
    v8bf t0 = *(const v8bf*)(Ag + (size_t)r0 * lda + s0);
    v8bf t1 = *(const v8bf*)(Ag + (size_t)(r0 + 64) * lda + s0);
    v8bf t2 = *(const v8bf*)(Wg + (size_t)r0 * ldw + s0);
    v8bf t3 = *(const v8bf*)(Wg + (size_t)(r0 + 64) * ldw + s0);
    *(v8bf*)&As[0][lo] = t0;
    *(v8bf*)&As[0][hi] = t1;
    *(v8bf*)&Bs[0][lo] = t2;
    *(v8bf*)&Bs[0][hi] = t3;
  }
#endif
  __syncthreads();

  int buf = 0;
  for (int ks = 0; ks < K; ks += 32) {
    const bool more = (ks + 32) < K;
#if USE_ASYNC_LDS
    if (more) stage(ks + 32, buf ^ 1);
#else
    v8bf t0, t1, t2, t3;
    if (more) {
      t0 = *(const v8bf*)(Ag + (size_t)r0 * lda + (ks + 32) + s0);
      t1 = *(const v8bf*)(Ag + (size_t)(r0 + 64) * lda + (ks + 32) + s0);
      t2 = *(const v8bf*)(Wg + (size_t)r0 * ldw + (ks + 32) + s0);
      t3 = *(const v8bf*)(Wg + (size_t)(r0 + 64) * ldw + (ks + 32) + s0);
    }
#endif
    // ---- compute from current LDS buffer
    {
      const __bf16* Ac = &As[buf][wm * 32 * LDS_LD];
      const __bf16* Bc = &Bs[buf][wn * 64 * LDS_LD];
      v16bf a0 = load_a_frag(Ac, LDS_LD, 0, lane);
      v16bf a1 = load_a_frag(Ac + 16 * LDS_LD, LDS_LD, 0, lane);
#pragma unroll
      for (int j = 0; j < 4; ++j) {
        v16bf bj = load_b_frag(Bc + j * 16 * LDS_LD, LDS_LD, 0, lane);
        acc0[j] = wmma_bf16(a0, bj, acc0[j]);
        acc1[j] = wmma_bf16(a1, bj, acc1[j]);
      }
    }
#if USE_ASYNC_LDS
    if (more) __builtin_amdgcn_s_wait_asynccnt(0);
#else
    if (more) {
      *(v8bf*)&As[buf ^ 1][lo] = t0;
      *(v8bf*)&As[buf ^ 1][hi] = t1;
      *(v8bf*)&Bs[buf ^ 1][lo] = t2;
      *(v8bf*)&Bs[buf ^ 1][hi] = t3;
    }
#endif
    __syncthreads();
    buf ^= 1;
  }

  // C/D layout: lane%16 = column, VGPR e = row e (+8 for lanes>=16)
  const int r  = m0 + wm * 32 + ((lane >> 4) << 3);
  const int cN = n0 + wn * 64 + (lane & 15);
#pragma unroll
  for (int e = 0; e < 8; ++e) {
#pragma unroll
    for (int j = 0; j < 4; ++j) {
      epi(r + e,      cN + j * 16, acc0[j][e]);
      epi(r + 16 + e, cN + j * 16, acc1[j][e]);
    }
  }
}

// ---------------------------------------------------------------------------
// Windowed attention: one block per (window, head); 4 waves, each owning
// 16 query rows. S = Q K^T (WMMA) -> scale + shift-mask + softmax in
// registers (xor-shuffles within the 16-lane C-layout halves) -> P in LDS,
// V^T staged in LDS -> O = P V (WMMA).
// ---------------------------------------------------------------------------
__global__ __launch_bounds__(128)
void attn_win(const __bf16* __restrict__ qkv, __bf16* __restrict__ aout) {
  __shared__ __bf16 Ps[64 * 64];
  __shared__ __bf16 Vt[64 * 64];
  const int w    = blockIdx.x;   // 0..1023  (global window = b*256 + local)
  const int hd   = blockIdx.y;   // 0..11
  const int lane = threadIdx.x & 31;
  const int wv   = threadIdx.x >> 5;

  const __bf16* Qb = qkv + (size_t)(w * 64) * C3 + hd * 64;
  const __bf16* Kb = Qb + 768;
  const __bf16* Vb = Qb + 1536;

  // stage V^T into LDS: Vt[d][j] = V[j][d]
  for (int idx = threadIdx.x; idx < 4096; idx += 128) {
    int j = idx >> 6, d = idx & 63;
    Vt[d * 64 + j] = Vb[(size_t)j * C3 + d];
  }

  // S = Q K^T for rows [16*wv, 16*wv+16)
  const __bf16* Arow = Qb + (size_t)(wv * 16) * C3;
  v8f sAcc[4] = {};
#pragma unroll
  for (int k = 0; k < 64; k += 32) {
    v16bf a = load_a_frag(Arow, C3, k, lane);
#pragma unroll
    for (int j = 0; j < 4; ++j)
      sAcc[j] = wmma_bf16(a, load_b_frag(Kb + (size_t)(j * 16) * C3, C3, k, lane), sAcc[j]);
  }

  // shifted-window mask groups (computed on the fly; warped merge: wr 1->2)
  const int wl = w & 255, wr = wl >> 4, wc = wl & 15;
  int gcol[4];
#pragma unroll
  for (int j = 0; j < 4; ++j) {
    int t  = j * 16 + (lane & 15);
    int hs = wr * 8 + (t >> 3), ws = wc * 8 + (t & 7);
    gcol[j] = 3 * (hs < 120 ? 0 : (hs < 124 ? 1 : 2)) + (ws >= 120 ? 2 : 0);
  }

  const int rbase = wv * 16 + ((lane >> 4) << 3);
#pragma unroll
  for (int e = 0; e < 8; ++e) {
    int rtok = rbase + e;
    int hs = wr * 8 + (rtok >> 3), ws = wc * 8 + (rtok & 7);
    int gr = 3 * (hs < 120 ? 0 : (hs < 124 ? 1 : 2)) + (ws >= 120 ? 2 : 0);
    float sv[4];
    float mx = -3.0e38f;
#pragma unroll
    for (int j = 0; j < 4; ++j) {
      sv[j] = sAcc[j][e] * 0.125f + (gr == gcol[j] ? 0.0f : -100.0f);
      mx = fmaxf(mx, sv[j]);
    }
#pragma unroll
    for (int m = 8; m >= 1; m >>= 1) mx = fmaxf(mx, __shfl_xor(mx, m, 32));
    float sum = 0.0f;
#pragma unroll
    for (int j = 0; j < 4; ++j) { sv[j] = __expf(sv[j] - mx); sum += sv[j]; }
#pragma unroll
    for (int m = 8; m >= 1; m >>= 1) sum += __shfl_xor(sum, m, 32);
    float inv = 1.0f / sum;
#pragma unroll
    for (int j = 0; j < 4; ++j)
      Ps[rtok * 64 + j * 16 + (lane & 15)] = (__bf16)(sv[j] * inv);
  }
  __syncthreads();

  // O = P V
  v8f oAcc[4] = {};
#pragma unroll
  for (int k = 0; k < 64; k += 32) {
    v16bf a = load_a_frag(Ps + wv * 16 * 64, 64, k, lane);
#pragma unroll
    for (int j = 0; j < 4; ++j)
      oAcc[j] = wmma_bf16(a, load_b_frag(Vt + j * 16 * 64, 64, k, lane), oAcc[j]);
  }

  __bf16* orow = aout + (size_t)(w * 64) * CCH + hd * 64;
#pragma unroll
  for (int e = 0; e < 8; ++e) {
    int rtok = rbase + e;
#pragma unroll
    for (int j = 0; j < 4; ++j)
      orow[(size_t)rtok * CCH + j * 16 + (lane & 15)] = (__bf16)oAcc[j][e];
  }
}

// ---------------------------------------------------------------------------
// LayerNorm (one wave32 per token, 24 channels/lane) with optional
// cyclic-shift + window-partition gather; output bf16.
// ---------------------------------------------------------------------------
__global__ __launch_bounds__(256)
void ln_to_bf16(const float* __restrict__ x, const float* __restrict__ gw,
                const float* __restrict__ gb, __bf16* __restrict__ out, int shifted) {
  const int tok  = blockIdx.x * 8 + (threadIdx.x >> 5);
  const int lane = threadIdx.x & 31;
  size_t src_tok;
  if (shifted) {
    int wdx = tok >> 6, t = tok & 63;
    int bi = wdx >> 8, wl = wdx & 255;
    int hs = (wl >> 4) * 8 + (t >> 3);
    int ws = (wl & 15) * 8 + (t & 7);
    int h  = (hs + 4) & 127, wp = (ws + 4) & 127;
    src_tok = (size_t)bi * 16384 + (size_t)h * 128 + wp;
  } else {
    src_tok = (size_t)tok;
  }
  const float* src = x + src_tok * CCH;
  float v[24];
  float s = 0.0f;
#pragma unroll
  for (int i = 0; i < 24; ++i) { v[i] = src[lane + 32 * i]; s += v[i]; }
#pragma unroll
  for (int m = 16; m >= 1; m >>= 1) s += __shfl_xor(s, m, 32);
  float mean = s * (1.0f / 768.0f);
  float q = 0.0f;
#pragma unroll
  for (int i = 0; i < 24; ++i) { float d = v[i] - mean; q += d * d; }
#pragma unroll
  for (int m = 16; m >= 1; m >>= 1) q += __shfl_xor(q, m, 32);
  float rstd = rsqrtf(q * (1.0f / 768.0f) + 1e-5f);
  __bf16* dst = out + (size_t)tok * CCH;
#pragma unroll
  for (int i = 0; i < 24; ++i) {
    int c = lane + 32 * i;
    dst[c] = (__bf16)((v[i] - mean) * rstd * gw[c] + gb[c]);
  }
}

// f32 -> bf16 weight conversion
__global__ __launch_bounds__(256)
void f2bf(const float* __restrict__ in, __bf16* __restrict__ out, int n) {
  int i = blockIdx.x * 256 + threadIdx.x;
  if (i < n) out[i] = (__bf16)in[i];
}

// ---------------------------------------------------------------------------
// Launch
// ---------------------------------------------------------------------------
extern "C" void kernel_launch(void* const* d_in, const int* in_sizes, int n_in,
                              void* d_out, int out_size, void* d_ws, size_t ws_size,
                              hipStream_t stream) {
  const float* x      = (const float*)d_in[0];
  const float* n1w    = (const float*)d_in[1];
  const float* n1b    = (const float*)d_in[2];
  const float* qkv_w  = (const float*)d_in[3];
  const float* qkv_b  = (const float*)d_in[4];
  const float* proj_w = (const float*)d_in[5];
  const float* proj_b = (const float*)d_in[6];
  const float* n2w    = (const float*)d_in[7];
  const float* n2b    = (const float*)d_in[8];
  const float* fc1_w  = (const float*)d_in[9];
  const float* fc1_b  = (const float*)d_in[10];
  const float* fc2_w  = (const float*)d_in[11];
  const float* fc2_b  = (const float*)d_in[12];
  float* out = (float*)d_out;
  char*  ws  = (char*)d_ws;

  // Workspace layout (regions reused across phases; total ~494 MB):
  //   [0,           301989888)  QKV (bf16 65536x2304); later YLN overlays [0, 100663296)
  //   [301989888,   402653184)  XW = LN1+window (bf16 65536x768); later attention output
  //   [100663296,   503316480)  HID (bf16 65536x3072) -- written after XW/QKV are dead
  //   [503316480,   ...)        bf16 weight copies (persist whole call)
  __bf16* QKV  = (__bf16*)(ws + 0);
  __bf16* XW   = (__bf16*)(ws + 301989888ull);
  __bf16* YLN  = (__bf16*)(ws + 0);
  __bf16* HID  = (__bf16*)(ws + 100663296ull);
  __bf16* QKVW = (__bf16*)(ws + 503316480ull);   // 2304*768
  __bf16* PRJW = (__bf16*)(ws + 506855424ull);   // 768*768
  __bf16* FC1W = (__bf16*)(ws + 508035072ull);   // 3072*768
  __bf16* FC2W = (__bf16*)(ws + 512753664ull);   // 768*3072

  f2bf<<<(2304 * 768 + 255) / 256, 256, 0, stream>>>(qkv_w, QKVW, 2304 * 768);
  f2bf<<<(768 * 768 + 255) / 256, 256, 0, stream>>>(proj_w, PRJW, 768 * 768);
  f2bf<<<(3072 * 768 + 255) / 256, 256, 0, stream>>>(fc1_w, FC1W, 3072 * 768);
  f2bf<<<(768 * 3072 + 255) / 256, 256, 0, stream>>>(fc2_w, FC2W, 768 * 3072);

  // 1) LN1 + shift + window partition -> XW (bf16, windowed order)
  ln_to_bf16<<<TOK_TOTAL / 8, 256, 0, stream>>>(x, n1w, n1b, XW, 1);

  // 2) QKV projection: 65536 x 2304 x 768
  gemm_bf16<EpiBias><<<dim3(TOK_TOTAL / 128, C3 / 128), 256, 0, stream>>>(
      XW, QKVW, CCH, CCH, CCH, EpiBias{qkv_b, QKV, C3});

  // 3) Windowed attention (writes attention output over the dead XW region)
  attn_win<<<dim3(1024, 12), 128, 0, stream>>>(QKV, XW);

  // 4) Output projection + window-reverse + unshift + residual -> d_out (f32)
  gemm_bf16<EpiProj><<<dim3(TOK_TOTAL / 128, CCH / 128), 256, 0, stream>>>(
      XW, PRJW, CCH, CCH, CCH, EpiProj{proj_b, x, out});

  // 5) LN2 -> YLN (bf16)
  ln_to_bf16<<<TOK_TOTAL / 8, 256, 0, stream>>>(out, n2w, n2b, YLN, 0);

  // 6) fc1 + exact GELU: 65536 x 3072 x 768
  gemm_bf16<EpiGelu><<<dim3(TOK_TOTAL / 128, CHID / 128), 256, 0, stream>>>(
      YLN, FC1W, CCH, CCH, CCH, EpiGelu{fc1_b, HID, CHID});

  // 7) fc2 + residual add into d_out: 65536 x 768 x 3072
  gemm_bf16<EpiAdd><<<dim3(TOK_TOTAL / 128, CCH / 128), 256, 0, stream>>>(
      HID, FC2W, CHID, CHID, CHID, EpiAdd{fc2_b, out});
}
